// GroupChannelAttention_81930796139038
// MI455X (gfx1250) — matrix operand order repaired and checked by reading, not compiled, not run on hardware
//
#include <hip/hip_runtime.h>
#include <hip/hip_bf16.h>
#include <math.h>

typedef __attribute__((ext_vector_type(8)))  _Float16 v8h;
typedef __attribute__((ext_vector_type(16))) _Float16 v16h;
typedef __attribute__((ext_vector_type(8)))  float    v8f;

#define DIMC 256
#define HEADS 4
#define HD 64
#define MLP 768
#define BB 8
#define NN 4096
#define BN (BB*NN)        // 32768 tokens
#define KSCALE 0.125f     // HEAD_DIM^-0.5

// ---------------------------------------------------------------------------
// WMMA fragment loaders (v_wmma_f32_16x16x32_f16, wave32; this code's contiguous k order, same for A and B).
// A (16x32 f16): lane L holds row M=L%16, K={kb+half*16 .. +15}:
//                one contiguous 32B load.
// B (32x16 f16): with B pre-transposed (Bt[N,K] row-major), lane c holds
//                column c, K={kb+half*16 .. +15}: one contiguous 32B load.
// C/D (16x16 f32): VGPR j holds D[half*8+j, c].
// ---------------------------------------------------------------------------
__device__ __forceinline__ v16h load_a_frag(const _Float16* __restrict__ A,
                                            int lda, int kb, int lane) {
  const int r = lane & 15, hf = lane >> 4;
  return *(const v16h*)(A + (size_t)r * lda + kb + hf * 16);
}

__device__ __forceinline__ v16h load_b_frag(const _Float16* __restrict__ Bt,
                                            int ldb, int kb, int lane) {
  const int c = lane & 15, hf = lane >> 4;
  return *(const v16h*)(Bt + (size_t)c * ldb + kb + hf * 16);
}

// One wave computes a 16 x (NT*16) strip; A fragment reused NT times per k-step.
template<int NT>
__device__ __forceinline__ void wmma_strip(const _Float16* __restrict__ A, int lda,
                                           const _Float16* __restrict__ Bt, int ldb,
                                           int K, int lane, v8f* acc) {
  for (int kb = 0; kb < K; kb += 32) {
    v16h a = load_a_frag(A, lda, kb, lane);
#pragma unroll
    for (int t = 0; t < NT; ++t) {
      v16h b = load_b_frag(Bt + (size_t)(t * 16) * ldb, ldb, kb, lane);
      acc[t] = __builtin_amdgcn_wmma_f32_16x16x32_f16(
          /*neg_a=*/false, a, /*neg_b=*/false, b,
          /*c_mod=*/(short)0, acc[t], /*reuse_a=*/false, /*reuse_b=*/false);
    }
  }
}

// ---------------------------------------------------------------------------
// Precision conversion / weight transpose
// ---------------------------------------------------------------------------
__global__ void __launch_bounds__(256)
k_cvt_f16(const float* __restrict__ s, _Float16* __restrict__ d, int n) {
  int i = blockIdx.x * blockDim.x + threadIdx.x;
  if (i < n) d[i] = (_Float16)s[i];
}

// WT[n*K + k] = W[k*Nc + n]  (W is [K,Nc] row-major -> WT is [Nc,K] row-major)
__global__ void __launch_bounds__(256)
k_transpose_cvt(const float* __restrict__ W, _Float16* __restrict__ WT,
                int K, int Nc) {
  int i = blockIdx.x * blockDim.x + threadIdx.x;
  if (i < K * Nc) {
    int n = i / K, k = i - n * K;
    WT[i] = (_Float16)W[(size_t)k * Nc + n];
  }
}

// ---------------------------------------------------------------------------
// QKV GEMM: [32768,256] x [256,768] + bias. Epilogue scatters into
//   q_f16 [B,H,N,64], kT_f16 [B,H,64,N] (x 0.125), vT_f16 [B,H,64,N]
// ---------------------------------------------------------------------------
__global__ void __launch_bounds__(256)
k_qkv_gemm(const _Float16* __restrict__ X, const _Float16* __restrict__ WT,
           const float* __restrict__ bias,
           _Float16* __restrict__ qf, _Float16* __restrict__ kT,
           _Float16* __restrict__ vT) {
  const int NS = MLP / 64;                       // 12 column strips
  int wave = (blockIdx.x * blockDim.x + threadIdx.x) >> 5;
  int lane = threadIdx.x & 31;
  if (wave >= (BN / 16) * NS) return;
  int mt = wave / NS, ns = wave - mt * NS;
  const _Float16* A  = X  + (size_t)mt * 16 * DIMC;
  const _Float16* Bt = WT + (size_t)ns * 64 * DIMC;
  v8f acc[4] = {};
  wmma_strip<4>(A, DIMC, Bt, DIMC, DIMC, lane, acc);

  const int c = lane & 15, hf = lane >> 4;
  const int row0 = mt * 16 + hf * 8;
#pragma unroll
  for (int t = 0; t < 4; ++t) {
    int cg = ns * 64 + t * 16 + c;               // column in [0,768)
    int s = cg >> 8, w = cg & 255, h = w >> 6, d = w & 63;
    float bi = bias[cg];
#pragma unroll
    for (int j = 0; j < 8; ++j) {
      int row = row0 + j;
      int b = row >> 12, n = row & (NN - 1);
      size_t bh = (size_t)(b * HEADS + h);
      float v = acc[t][j] + bi;
      if (s == 0)      qf[(bh * NN + n) * HD + d] = (_Float16)v;
      else if (s == 1) kT[(bh * HD + d) * NN + n] = (_Float16)(v * KSCALE);
      else             vT[(bh * HD + d) * NN + n] = (_Float16)v;
    }
  }
}

// ---------------------------------------------------------------------------
// attn_raw[bh][d][e] = sum_n kT[d,n] * vT[e,n]   (M=64, N=64, K=4096 per bh)
// ---------------------------------------------------------------------------
__global__ void __launch_bounds__(256)
k_attn_gemm(const _Float16* __restrict__ kT, const _Float16* __restrict__ vT,
            float* __restrict__ attn_raw) {
  int wave = (blockIdx.x * blockDim.x + threadIdx.x) >> 5;   // 128 waves
  int lane = threadIdx.x & 31;
  if (wave >= BB * HEADS * 4) return;
  int bh = wave >> 2, mt = wave & 3;
  const _Float16* A  = kT + (size_t)bh * HD * NN + (size_t)mt * 16 * NN;
  const _Float16* Bt = vT + (size_t)bh * HD * NN;
  v8f acc[4] = {};
  wmma_strip<4>(A, NN, Bt, NN, NN, lane, acc);

  const int c = lane & 15, hf = lane >> 4;
#pragma unroll
  for (int t = 0; t < 4; ++t)
#pragma unroll
    for (int j = 0; j < 8; ++j)
      attn_raw[(size_t)bh * HD * HD + (size_t)(mt * 16 + hf * 8 + j) * HD +
               t * 16 + c] = acc[t][j];
}

// ---------------------------------------------------------------------------
// Row softmax over e (64 elems), output f16 in natural [d,e] layout
// (which is exactly the Bt layout the xf GEMM needs).
// ---------------------------------------------------------------------------
__global__ void __launch_bounds__(64)
k_softmax64(const float* __restrict__ attn_raw, _Float16* __restrict__ attn_h) {
  __shared__ float s[64];
  int row = blockIdx.x, e = threadIdx.x;
  float v = attn_raw[(size_t)row * 64 + e];
  s[e] = v; __syncthreads();
  for (int off = 32; off > 0; off >>= 1) {
    if (e < off) s[e] = fmaxf(s[e], s[e + off]);
    __syncthreads();
  }
  float mx = s[0]; __syncthreads();
  float ex = expf(v - mx);
  s[e] = ex; __syncthreads();
  for (int off = 32; off > 0; off >>= 1) {
    if (e < off) s[e] += s[e + off];
    __syncthreads();
  }
  attn_h[(size_t)row * 64 + e] = (_Float16)(ex / s[0]);
}

// ---------------------------------------------------------------------------
// xf[n,d] = sum_e q[n,e]*attn[d,e]; x1 = x + xf  (M=4096, N=64, K=64 per bh)
// ---------------------------------------------------------------------------
__global__ void __launch_bounds__(256)
k_xf_gemm(const _Float16* __restrict__ qf, const _Float16* __restrict__ attn_h,
          const float* __restrict__ x, float* __restrict__ x1) {
  int wave = (blockIdx.x * blockDim.x + threadIdx.x) >> 5;   // 8192 waves
  int lane = threadIdx.x & 31;
  if (wave >= BB * HEADS * (NN / 16)) return;
  int bh = wave >> 8, mt = wave & 255;
  const _Float16* A  = qf + (size_t)bh * NN * HD + (size_t)mt * 16 * HD;
  const _Float16* Bt = attn_h + (size_t)bh * HD * HD;
  v8f acc[4] = {};
  wmma_strip<4>(A, HD, Bt, HD, HD, lane, acc);

  int b = bh >> 2, h = bh & 3;
  const int c = lane & 15, hf = lane >> 4;
#pragma unroll
  for (int t = 0; t < 4; ++t)
#pragma unroll
    for (int j = 0; j < 8; ++j) {
      int n = mt * 16 + hf * 8 + j;
      int col = h * HD + t * 16 + c;
      size_t idx = ((size_t)b * NN + n) * DIMC + col;
      x1[idx] = x[idx] + acc[t][j];
    }
}

// ---------------------------------------------------------------------------
// LayerNorm over DIM=256 per token -> f16
// ---------------------------------------------------------------------------
__global__ void __launch_bounds__(256)
k_layernorm(const float* __restrict__ x1, const float* __restrict__ g,
            const float* __restrict__ be, _Float16* __restrict__ ln) {
  __shared__ float s[DIMC];
  size_t tok = blockIdx.x;
  int c = threadIdx.x;
  float v = x1[tok * DIMC + c];
  s[c] = v; __syncthreads();
  for (int off = 128; off > 0; off >>= 1) {
    if (c < off) s[c] += s[c + off];
    __syncthreads();
  }
  float mu = s[0] * (1.0f / DIMC); __syncthreads();
  float dv = v - mu;
  s[c] = dv * dv; __syncthreads();
  for (int off = 128; off > 0; off >>= 1) {
    if (c < off) s[c] += s[c + off];
    __syncthreads();
  }
  float r = rsqrtf(s[0] * (1.0f / DIMC) + 1e-5f);
  ln[tok * DIMC + c] = (_Float16)(dv * r * g[c] + be[c]);
}

// ---------------------------------------------------------------------------
// FC1: [32768,256] x [256,768] + bias, exact GELU, -> f16
// ---------------------------------------------------------------------------
__global__ void __launch_bounds__(256)
k_fc1_gemm(const _Float16* __restrict__ LN, const _Float16* __restrict__ WT,
           const float* __restrict__ bias, _Float16* __restrict__ hid) {
  const int NS = MLP / 64;
  int wave = (blockIdx.x * blockDim.x + threadIdx.x) >> 5;
  int lane = threadIdx.x & 31;
  if (wave >= (BN / 16) * NS) return;
  int mt = wave / NS, ns = wave - mt * NS;
  const _Float16* A  = LN + (size_t)mt * 16 * DIMC;
  const _Float16* Bt = WT + (size_t)ns * 64 * DIMC;
  v8f acc[4] = {};
  wmma_strip<4>(A, DIMC, Bt, DIMC, DIMC, lane, acc);

  const int c = lane & 15, hf = lane >> 4;
  const int row0 = mt * 16 + hf * 8;
#pragma unroll
  for (int t = 0; t < 4; ++t) {
    int cg = ns * 64 + t * 16 + c;
    float bi = bias[cg];
#pragma unroll
    for (int j = 0; j < 8; ++j) {
      float v = acc[t][j] + bi;
      float ge = 0.5f * v * (1.0f + erff(v * 0.70710678118654752f));  // exact GELU
      hid[(size_t)(row0 + j) * MLP + cg] = (_Float16)ge;
    }
  }
}

// ---------------------------------------------------------------------------
// FC2: [32768,768] x [768,256] + bias + residual(x1) -> out (fp32)
// ---------------------------------------------------------------------------
__global__ void __launch_bounds__(256)
k_fc2_gemm(const _Float16* __restrict__ HID, const _Float16* __restrict__ WT,
           const float* __restrict__ bias, const float* __restrict__ x1,
           float* __restrict__ out) {
  const int NS = DIMC / 64;                      // 4 strips
  int wave = (blockIdx.x * blockDim.x + threadIdx.x) >> 5;
  int lane = threadIdx.x & 31;
  if (wave >= (BN / 16) * NS) return;
  int mt = wave / NS, ns = wave - mt * NS;
  const _Float16* A  = HID + (size_t)mt * 16 * MLP;
  const _Float16* Bt = WT  + (size_t)ns * 64 * MLP;
  v8f acc[4] = {};
  wmma_strip<4>(A, MLP, Bt, MLP, MLP, lane, acc);

  const int c = lane & 15, hf = lane >> 4;
  const int row0 = mt * 16 + hf * 8;
#pragma unroll
  for (int t = 0; t < 4; ++t) {
    int cg = ns * 64 + t * 16 + c;
    float bi = bias[cg];
#pragma unroll
    for (int j = 0; j < 8; ++j) {
      size_t idx = (size_t)(row0 + j) * DIMC + cg;
      out[idx] = x1[idx] + acc[t][j] + bi;
    }
  }
}

// ---------------------------------------------------------------------------
// Launch: ~98 MB workspace, regions reused across phases.
// ---------------------------------------------------------------------------
extern "C" void kernel_launch(void* const* d_in, const int* in_sizes, int n_in,
                              void* d_out, int out_size, void* d_ws, size_t ws_size,
                              hipStream_t stream) {
  const float* x     = (const float*)d_in[0];
  const float* w_qkv = (const float*)d_in[1];
  const float* b_qkv = (const float*)d_in[2];
  const float* ng    = (const float*)d_in[3];
  const float* nb    = (const float*)d_in[4];
  const float* w_fc1 = (const float*)d_in[5];
  const float* b_fc1 = (const float*)d_in[6];
  const float* w_fc2 = (const float*)d_in[7];
  const float* b_fc2 = (const float*)d_in[8];
  float* out = (float*)d_out;

  char* ws = (char*)d_ws;
  // [0, 16 MiB):   x_f16, later reused as ln_f16
  _Float16* x_f16   = (_Float16*)(ws);
  _Float16* ln_f16  = x_f16;
  // [16 MiB, 64 MiB): q/kT/vT, later reused as hid_f16 (exactly 48 MiB)
  _Float16* q_f16   = (_Float16*)(ws + 16777216ull);
  _Float16* kT_f16  = (_Float16*)(ws + 33554432ull);
  _Float16* vT_f16  = (_Float16*)(ws + 50331648ull);
  _Float16* hid_f16 = q_f16;
  float*    x1      = (float*)(ws + 67108864ull);      // 32 MiB fp32 residual
  float*    attn_r  = (float*)(ws + 100663296ull);     // 512 KiB
  _Float16* attn_h  = (_Float16*)(ws + 101187584ull);  // 256 KiB
  _Float16* wqkvT   = (_Float16*)(ws + 101449728ull);  // 384 KiB each
  _Float16* wfc1T   = (_Float16*)(ws + 101842944ull);
  _Float16* wfc2T   = (_Float16*)(ws + 102236160ull);

  // 0) precision conversion + weight transposes (tiny, L2-resident afterwards)
  k_cvt_f16<<<(BN * DIMC) / 256, 256, 0, stream>>>(x, x_f16, BN * DIMC);
  k_transpose_cvt<<<(DIMC * MLP) / 256, 256, 0, stream>>>(w_qkv, wqkvT, DIMC, MLP);
  k_transpose_cvt<<<(DIMC * MLP) / 256, 256, 0, stream>>>(w_fc1, wfc1T, DIMC, MLP);
  k_transpose_cvt<<<(MLP * DIMC) / 256, 256, 0, stream>>>(w_fc2, wfc2T, MLP, DIMC);

  // 1) QKV projection (2048 M-tiles x 12 strips = 24576 waves)
  k_qkv_gemm<<<3072, 256, 0, stream>>>(x_f16, wqkvT, b_qkv, q_f16, kT_f16, vT_f16);

  // 2) channel attention K^T V (128 waves)
  k_attn_gemm<<<16, 256, 0, stream>>>(kT_f16, vT_f16, attn_r);

  // 3) softmax over e (2048 rows)
  k_softmax64<<<BB * HEADS * HD, 64, 0, stream>>>(attn_r, attn_h);

  // 4) xf = q @ attn^T + residual (8192 waves)
  k_xf_gemm<<<1024, 256, 0, stream>>>(q_f16, attn_h, x, x1);

  // 5) LayerNorm
  k_layernorm<<<BN, 256, 0, stream>>>(x1, ng, nb, ln_f16);

  // 6) FC1 + exact GELU (24576 waves)
  k_fc1_gemm<<<3072, 256, 0, stream>>>(ln_f16, wfc1T, b_fc1, hid_f16);

  // 7) FC2 + bias + residual (8192 waves)
  k_fc2_gemm<<<1024, 256, 0, stream>>>(hid_f16, wfc2T, b_fc2, x1, out);

  (void)in_sizes; (void)n_in; (void)out_size; (void)ws_size;
}
